// StreamableLSTM_78039555768917
// MI455X (gfx1250) — matrix operand order repaired and checked by reading, not compiled, not run on hardware
//
#include <hip/hip_runtime.h>
#include <hip/hip_bf16.h>
#include <math.h>

// ---------------------------------------------------------------------------
// StreamableLSTM for MI455X (gfx1250): 2-layer LSTM, D=512, B=32, T=1500.
// Phase 1: xT = transpose+bf16(x)                    (parallel)
// Phase 2: xW = xT @ Wih0^T + b0   (bf16 WMMA GEMM)  (parallel)
// Phase 3: layer-0 recurrence, persistent kernel, grid barrier per step
// Phase 4: xW = h0_seq @ Wih1^T + b1                 (parallel)
// Phase 5: layer-1 recurrence, writes out = h1 + x directly
// ---------------------------------------------------------------------------

typedef __attribute__((ext_vector_type(8)))  __bf16 bf16x8;
typedef __attribute__((ext_vector_type(16))) __bf16 v16bf;
typedef __attribute__((ext_vector_type(8)))  float  v8f;

#define DH   512
#define BB   32
#define GG   2048          // 4*DH
#define TT   1500
#define NBLK 32            // persistent recurrence workgroups (must co-reside)

// float -> bf16 bits, round-to-nearest-even
__device__ __forceinline__ unsigned short f2bf(float f) {
  union { float f; unsigned u; } v; v.f = f;
  unsigned r = v.u + 0x7FFFu + ((v.u >> 16) & 1u);
  return (unsigned short)(r >> 16);
}

__device__ __forceinline__ bf16x8 ld8(const unsigned short* p) {
  return *reinterpret_cast<const bf16x8*>(p);
}

__device__ __forceinline__ v16bf cat16(bf16x8 lo, bf16x8 hi) {
  return __builtin_shufflevector(lo, hi, 0,1,2,3,4,5,6,7,8,9,10,11,12,13,14,15);
}

__device__ __forceinline__ v8f wmma_bf16(v16bf a, v16bf b, v8f c) {
  return __builtin_amdgcn_wmma_f32_16x16x32_bf16(
      /*neg_a=*/false, a, /*neg_b=*/false, b,
      /*c_mod=*/(short)0, c, /*reuse_a=*/false, /*reuse_b=*/false);
}

// --- Async global -> LDS copy (CDNA5 GLOBAL_LOAD_ASYNC_TO_LDS_B128) --------
// Probe result: builtin exists and takes (int4 addrspace(1)*, int4
// addrspace(3)*, imm offset, imm cpol).
#if __has_builtin(__builtin_amdgcn_global_load_async_to_lds_b128)
#define HAS_ASYNC_LDS 1
#else
#define HAS_ASYNC_LDS 0
#endif

typedef int v4i_vs __attribute__((__vector_size__(16)));
typedef __attribute__((address_space(1))) v4i_vs* g_v4i_ptr;
typedef __attribute__((address_space(3))) v4i_vs* l_v4i_ptr;

__device__ __forceinline__ void cp_g2l_16B(const unsigned short* g,
                                           unsigned short* l) {
#if HAS_ASYNC_LDS
  __builtin_amdgcn_global_load_async_to_lds_b128(
      (g_v4i_ptr)(void*)g, (l_v4i_ptr)(void*)l, /*offset=*/0, /*cpol=*/0);
#else
  *reinterpret_cast<bf16x8*>(l) = ld8(g);
#endif
}

__device__ __forceinline__ void cp_g2l_wait() {
#if HAS_ASYNC_LDS
#if __has_builtin(__builtin_amdgcn_s_wait_asynccnt)
  __builtin_amdgcn_s_wait_asynccnt(0);
#else
  asm volatile("s_wait_asynccnt 0x0" ::: "memory");
#endif
#endif
}

// Invalidate WGP-level caches so the next step's h broadcast is read fresh
// from L2 (device-coherent). GLOBAL_INV tracked with LOADcnt.
__device__ __forceinline__ void inv_l0() {
  asm volatile("global_inv scope:SCOPE_DEV\n\t"
               "s_wait_loadcnt 0x0" ::: "memory");
}

// Sense-reversal grid barrier (agent scope). NBLK workgroups, all resident.
__device__ __forceinline__ void grid_sync(unsigned* sync) {
  __threadfence();      // each thread: flush its global stores to device scope
  __syncthreads();
  if (threadIdx.x == 0) {
    unsigned* cnt = sync;
    unsigned* gen = sync + 1;
    unsigned g = __hip_atomic_load(gen, __ATOMIC_ACQUIRE, __HIP_MEMORY_SCOPE_AGENT);
    unsigned t = __hip_atomic_fetch_add(cnt, 1u, __ATOMIC_ACQ_REL, __HIP_MEMORY_SCOPE_AGENT);
    if (t == NBLK - 1u) {
      __hip_atomic_store(cnt, 0u, __ATOMIC_RELAXED, __HIP_MEMORY_SCOPE_AGENT);
      __hip_atomic_fetch_add(gen, 1u, __ATOMIC_RELEASE, __HIP_MEMORY_SCOPE_AGENT);
    } else {
      while (__hip_atomic_load(gen, __ATOMIC_ACQUIRE, __HIP_MEMORY_SCOPE_AGENT) == g)
        __builtin_amdgcn_s_sleep(2);
    }
  }
  __syncthreads();
}

// ---------------------------------------------------------------------------
// Prep: fp32 weights -> bf16 ([4D, D] row-major kept: B[k,n] = W[n*D+k]),
// combined biases, zero barrier counter.
// ---------------------------------------------------------------------------
__global__ void lstm_prep_kernel(
    const float* __restrict__ Wih0, const float* __restrict__ Whh0,
    const float* __restrict__ Wih1, const float* __restrict__ Whh1,
    const float* __restrict__ bih0, const float* __restrict__ bhh0,
    const float* __restrict__ bih1, const float* __restrict__ bhh1,
    unsigned short* __restrict__ wih0b, unsigned short* __restrict__ whh0b,
    unsigned short* __restrict__ wih1b, unsigned short* __restrict__ whh1b,
    float* __restrict__ bias0, float* __restrict__ bias1,
    unsigned* __restrict__ sync) {
  size_t i = (size_t)blockIdx.x * blockDim.x + threadIdx.x;
  const size_t WN = (size_t)GG * DH;
  if (i < WN) {
    wih0b[i] = f2bf(Wih0[i]);
    whh0b[i] = f2bf(Whh0[i]);
    wih1b[i] = f2bf(Wih1[i]);
    whh1b[i] = f2bf(Whh1[i]);
  }
  if (i < GG) {
    bias0[i] = bih0[i] + bhh0[i];
    bias1[i] = bih1[i] + bhh1[i];
  }
  if (i == 0) sync[0] = 0u;   // barrier count; gen may hold any value
}

// ---------------------------------------------------------------------------
// x [B, D, T] fp32 -> xT [(t*B+b)*D + d] bf16  (coalesced writes)
// ---------------------------------------------------------------------------
__global__ void lstm_xpose_kernel(const float* __restrict__ x,
                                  unsigned short* __restrict__ xT) {
  size_t i = (size_t)blockIdx.x * blockDim.x + threadIdx.x;
  if (i >= (size_t)TT * BB * DH) return;
  int d = (int)(i & (DH - 1));
  size_t r = i >> 9;            // / DH
  int b = (int)(r & (BB - 1));
  int t = (int)(r >> 5);        // / BB
  xT[i] = f2bf(x[(size_t)b * DH * TT + (size_t)d * TT + t]);
}

// ---------------------------------------------------------------------------
// C[M,N] = A[M,K](bf16) @ W[N,K]^T(bf16) + bias[N]   (fp32 out)
// Block: 256 thr = 8 waves, block tile 128M x 128N, wave tile 32M x 64N,
// K-step 32 via v_wmma_f32_16x16x32_bf16.  M%128==0, N%128==0, K%32==0.
// ---------------------------------------------------------------------------
__global__ __launch_bounds__(256) void lstm_gemm_kernel(
    const unsigned short* __restrict__ A,
    const unsigned short* __restrict__ W,
    const float* __restrict__ bias,
    float* __restrict__ C,
    int M, int N, int K) {
  const int lane = threadIdx.x & 31;
  const int w    = threadIdx.x >> 5;
  const int mw   = w & 3, nw = w >> 2;
  const int m0   = blockIdx.y * 128 + mw * 32;
  const int n0   = blockIdx.x * 128 + nw * 64;
  const int lm   = lane & 15;     // A: tile row m; B/C: tile col n
  const int lh   = lane >> 4;     // lane half
  const int kbA  = lh * 8;        // A per-lane K group base (ISA 7.12.2)
  const int khB  = lh * 8;        // B per-lane K group base

  v8f acc[2][4];
#pragma unroll
  for (int i = 0; i < 2; ++i)
#pragma unroll
    for (int j = 0; j < 4; ++j)
      acc[i][j] = v8f{0.f, 0.f, 0.f, 0.f, 0.f, 0.f, 0.f, 0.f};

  for (int kk = 0; kk < K; kk += 32) {
    v16bf a[2], b[4];
#pragma unroll
    for (int mt = 0; mt < 2; ++mt) {
      const unsigned short* pa = A + (size_t)(m0 + mt * 16 + lm) * K + kk + kbA;
      a[mt] = cat16(ld8(pa), ld8(pa + 16));
    }
#pragma unroll
    for (int nt = 0; nt < 4; ++nt) {
      const unsigned short* pb = W + (size_t)(n0 + nt * 16 + lm) * K + kk + khB;
      b[nt] = cat16(ld8(pb), ld8(pb + 16));
    }
    if (kk + 32 < K)
      __builtin_prefetch(A + (size_t)(m0 + lm) * K + kk + 32, 0, 0);
#pragma unroll
    for (int mt = 0; mt < 2; ++mt)
#pragma unroll
      for (int nt = 0; nt < 4; ++nt)
        acc[mt][nt] = wmma_bf16(a[mt], b[nt], acc[mt][nt]);
  }

#pragma unroll
  for (int nt = 0; nt < 4; ++nt) {
    const int n  = n0 + nt * 16 + lm;
    const float bv = bias[n];
#pragma unroll
    for (int mt = 0; mt < 2; ++mt)
#pragma unroll
      for (int r = 0; r < 8; ++r) {
        const int m = m0 + mt * 16 + lh * 8 + r;
        C[(size_t)m * N + n] = acc[mt][nt][r] + bv;
      }
  }
}

// ---------------------------------------------------------------------------
// Persistent LSTM recurrence. NBLK=32 workgroups x 256 threads.
// Block owns 16 h-columns (jb..jb+15) -> 64 gate columns across i,f,g,o.
// Per step: gates = xW[t] + h_{t-1} @ Whh^T  (bf16 WMMA, Whh slice in LDS),
// LSTM cell in fp32 (c kept in LDS), h_t broadcast via L2 ping-pong buffer,
// atomic grid barrier + global_inv between steps. Staging uses async
// global->LDS DMA (ASYNCcnt) when the toolchain exposes it.
// ---------------------------------------------------------------------------
__global__ __launch_bounds__(256) void lstm_recur_kernel(
    const float* __restrict__ xW,            // [T*B, 4D] fp32
    const unsigned short* __restrict__ Whh,  // [4D, D] bf16
    unsigned short* __restrict__ hbuf,       // [2, B, D] bf16 ping-pong
    unsigned* __restrict__ sync,
    unsigned short* __restrict__ hseq,       // [T*B, D] bf16 out (layer 0) or null
    const float* __restrict__ xin,           // [B, D, T] (layer 1) or null
    float* __restrict__ outp,                // [B, D, T] (layer 1) or null
    int T) {
  __shared__ unsigned short whhs[64 * DH];   // 64 KB: Whh slice, persistent
  __shared__ unsigned short hA[BB * DH];     // 32 KB: h_{t-1} stage
  __shared__ float gatesS[4][BB][16];        //  8 KB
  __shared__ float cs[BB][16];               //  2 KB: cell state

  const int tid   = threadIdx.x;
  const int lane  = tid & 31;
  const int w     = tid >> 5;
  const int gate  = w & 3;        // 0:i 1:f 2:g 3:o
  const int mtile = w >> 2;       // 0..1 (B rows 0-15 / 16-31)
  const int jb    = blockIdx.x * 16;
  const int lm    = lane & 15;
  const int lh    = lane >> 4;

  // Stage this block's Whh slice via async DMA:
  // local row nl = gate*16 + j  -> [64][512]
  for (int v = tid; v < 64 * DH / 8; v += 256) {
    const int nl = v >> 6;              // 64 vec8 per 512-col row
    const int k8 = (v & 63) * 8;
    const int g  = nl >> 4, j = nl & 15;
    cp_g2l_16B(Whh + (size_t)(g * DH + jb + j) * DH + k8,
               &whhs[nl * DH + k8]);
  }
  // Zero this block's h columns in both ping-pong buffers, and cell state.
  for (int v = tid; v < 2 * BB * 16; v += 256) {
    const int par = v >> 9;             // / (BB*16)
    const int rb  = (v >> 4) & 31;
    const int j   = v & 15;
    hbuf[(size_t)par * BB * DH + rb * DH + jb + j] = 0;
  }
  for (int v = tid; v < BB * 16; v += 256) cs[v >> 4][v & 15] = 0.f;
  cp_g2l_wait();
  __syncthreads();
  grid_sync(sync);
  inv_l0();

  for (int t = 0; t < T; ++t) {
    // Stage full h_{t-1} (all 512 columns) into LDS via async DMA.
    const unsigned short* hr = hbuf + (size_t)(t & 1) * BB * DH;
    for (int v = tid; v < BB * DH / 8; v += 256)
      cp_g2l_16B(hr + v * 8, &hA[v * 8]);

    // While the DMA flies, init accumulator from precomputed xW slice
    // (includes both biases).
    const int ng = gate * DH + jb;      // global gate-column base for wave
    v8f acc0, acc1;
#pragma unroll
    for (int r = 0; r < 8; ++r) {
      const int brow = mtile * 16 + lh * 8 + r;
      acc0[r] = xW[(size_t)(t * BB + brow) * GG + ng + lm];
      acc1[r] = 0.f;
    }
    cp_g2l_wait();
    __syncthreads();

    // Recurrent GEMM: two interleaved accumulators halve the WMMA RAW chain.
#pragma unroll
    for (int kk = 0; kk < DH; kk += 64) {
      const unsigned short* pa0 = &hA[(mtile * 16 + lm) * DH + kk + lh * 8];
      const unsigned short* pb0 = &whhs[(gate * 16 + lm) * DH + kk + lh * 8];
      v16bf a0 = cat16(*reinterpret_cast<const bf16x8*>(pa0),
                       *reinterpret_cast<const bf16x8*>(pa0 + 16));
      v16bf b0 = cat16(*reinterpret_cast<const bf16x8*>(pb0),
                       *reinterpret_cast<const bf16x8*>(pb0 + 16));
      v16bf a1 = cat16(*reinterpret_cast<const bf16x8*>(pa0 + 32),
                       *reinterpret_cast<const bf16x8*>(pa0 + 48));
      v16bf b1 = cat16(*reinterpret_cast<const bf16x8*>(pb0 + 32),
                       *reinterpret_cast<const bf16x8*>(pb0 + 48));
      acc0 = wmma_bf16(a0, b0, acc0);
      acc1 = wmma_bf16(a1, b1, acc1);
    }
#pragma unroll
    for (int r = 0; r < 8; ++r)
      gatesS[gate][mtile * 16 + lh * 8 + r][lm] = acc0[r] + acc1[r];
    __syncthreads();

    // Prefetch next step's xW slice for this block (4 gates x 32 rows).
    if (t + 1 < T) {
      for (int e = tid; e < BB * 4; e += 256) {
        const int b = e & 31, g = e >> 5;
        __builtin_prefetch(&xW[(size_t)((t + 1) * BB + b) * GG + g * DH + jb], 0, 0);
      }
    }

    // Elementwise LSTM cell (fp32): 512 elements over 256 threads.
    for (int e = tid; e < BB * 16; e += 256) {
      const int b = e >> 4, j = e & 15;
      const float gi = gatesS[0][b][j];
      const float gf = gatesS[1][b][j];
      const float gg = gatesS[2][b][j];
      const float go = gatesS[3][b][j];
      const float si = 1.f / (1.f + __expf(-gi));
      const float sf = 1.f / (1.f + __expf(-gf));
      const float tg = tanhf(gg);
      const float so = 1.f / (1.f + __expf(-go));
      const float c  = sf * cs[b][j] + si * tg;
      cs[b][j] = c;
      const float h = so * tanhf(c);
      hbuf[(size_t)((t + 1) & 1) * BB * DH + b * DH + jb + j] = f2bf(h);
      if (hseq) hseq[(size_t)(t * BB + b) * DH + jb + j] = f2bf(h);
      if (outp) {
        const size_t oi = (size_t)b * DH * T + (size_t)(jb + j) * T + t;
        outp[oi] = h + xin[oi];   // skip connection fused into output
      }
    }
    grid_sync(sync);
    inv_l0();
  }
}

// ---------------------------------------------------------------------------
extern "C" void kernel_launch(void* const* d_in, const int* in_sizes, int n_in,
                              void* d_out, int out_size, void* d_ws, size_t ws_size,
                              hipStream_t stream) {
  (void)in_sizes; (void)n_in; (void)out_size; (void)ws_size;
  const float* x    = (const float*)d_in[0];
  const float* Wih0 = (const float*)d_in[1];
  const float* Whh0 = (const float*)d_in[2];
  const float* bih0 = (const float*)d_in[3];
  const float* bhh0 = (const float*)d_in[4];
  const float* Wih1 = (const float*)d_in[5];
  const float* Whh1 = (const float*)d_in[6];
  const float* bih1 = (const float*)d_in[7];
  const float* bhh1 = (const float*)d_in[8];

  const size_t WN = (size_t)GG * DH;       // 2048*512 weight elements
  const size_t MM = (size_t)TT * BB;       // 48000 rows

  char* ws = (char*)d_ws;
  size_t off = 0;
  auto alloc = [&](size_t bytes) -> void* {
    void* p = ws + off;
    off = (off + bytes + 255) & ~(size_t)255;
    return p;
  };
  unsigned short* wih0b = (unsigned short*)alloc(WN * 2);
  unsigned short* whh0b = (unsigned short*)alloc(WN * 2);
  unsigned short* wih1b = (unsigned short*)alloc(WN * 2);
  unsigned short* whh1b = (unsigned short*)alloc(WN * 2);
  float*          bias0 = (float*)alloc(GG * 4);
  float*          bias1 = (float*)alloc(GG * 4);
  unsigned*       syncp = (unsigned*)alloc(256);
  unsigned short* hbuf  = (unsigned short*)alloc(2 * BB * DH * 2);
  unsigned short* hseq  = (unsigned short*)alloc(MM * DH * 2); // xT, then h0_seq
  float*          xW    = (float*)alloc(MM * GG * 4);

  // 1) Weights/biases to bf16, reset barrier counter.
  lstm_prep_kernel<<<(unsigned)((WN + 255) / 256), 256, 0, stream>>>(
      Wih0, Whh0, Wih1, Whh1, bih0, bhh0, bih1, bhh1,
      wih0b, whh0b, wih1b, whh1b, bias0, bias1, syncp);

  // 2) xT = bf16 transpose of x  (stored in hseq buffer).
  lstm_xpose_kernel<<<(unsigned)((MM * DH + 255) / 256), 256, 0, stream>>>(x, hseq);

  // 3) xW = xT @ Wih0^T + (bih0+bhh0)
  dim3 ggrid(GG / 128, (unsigned)(MM / 128));
  lstm_gemm_kernel<<<ggrid, 256, 0, stream>>>(hseq, wih0b, bias0, xW,
                                              (int)MM, GG, DH);

  // 4) Layer-0 recurrence: writes h0_seq (bf16) over the consumed xT buffer.
  lstm_recur_kernel<<<NBLK, 256, 0, stream>>>(xW, whh0b, hbuf, syncp,
                                              hseq, nullptr, nullptr, TT);

  // 5) xW = h0_seq @ Wih1^T + (bih1+bhh1)
  lstm_gemm_kernel<<<ggrid, 256, 0, stream>>>(hseq, wih1b, bias1, xW,
                                              (int)MM, GG, DH);

  // 6) Layer-1 recurrence: out[b,d,t] = h1[t,b,d] + x[b,d,t] fused.
  lstm_recur_kernel<<<NBLK, 256, 0, stream>>>(xW, whh1b, hbuf, syncp,
                                              nullptr, x, (float*)d_out, TT);
}
